// LBONorm_88768384074407
// MI455X (gfx1250) — matrix operand order repaired and checked by reading, not compiled, not run on hardware
//
#include <hip/hip_runtime.h>

// ---------------------------------------------------------------------------
// LBONorm for MI455X (gfx1250, wave32).
// LayerNorm + rank-8 Gram projection + curvature energy + smoothed output.
// Bandwidth-bound (~256 MiB HBM @ 23.3 TB/s => ~11us floor); the 16x2048 x
// 2048x16 coefficient matmul runs on v_wmma_f32_16x16x32_bf16.
// Rank is padded 8->16 by DUPLICATING V rows (C cols 8..15 are never read),
// which avoids any zero-select VALU work in the B-fragment build.
// ---------------------------------------------------------------------------

typedef __attribute__((ext_vector_type(16))) __bf16 bf16x16;
typedef __attribute__((ext_vector_type(8)))  __bf16 bf16x8;
typedef __attribute__((ext_vector_type(8)))  float  f32x8;

constexpr int   kD       = 2048;
constexpr int   kRows    = 16;           // rows per workgroup (WMMA M)
constexpr int   kRank    = 8;            // low-rank dimension
constexpr int   kBlock   = 256;          // 8 waves
constexpr int   kWaves   = kBlock / 32;
constexpr int   kAStride = kD + 8;       // bf16 elems; +16B rotates LDS banks
constexpr float kEps     = 1e-5f;

__global__ __launch_bounds__(kBlock) void lbonorm_kernel(
    const float* __restrict__ h, const float* __restrict__ gamma,
    const float* __restrict__ beta, const float* __restrict__ V,
    const float* __restrict__ eta, float* __restrict__ out,
    float* __restrict__ eout) {
  __shared__ __bf16 Ahat[kRows * kAStride];   // ~64.3 KB: h_hat tile (bf16)
  __shared__ float  Cpart[kWaves][16][16];    // 8 KB: per-wave partial C
  __shared__ float  sCoef[kRows][16];         // final coefficients
  __shared__ float  sMu[kRows], sRs[kRows];
  __shared__ float  Epart[kRows][kWaves];     // per-wave e_curv partials

  const int tid     = threadIdx.x;
  const int wave    = tid >> 5;
  const int lane    = tid & 31;
  const int rowbase = blockIdx.x * kRows;
  const float etac  = fminf(fmaxf(eta[0], 0.0f), 0.5f);

  // ----- pass 1: each wave owns 2 rows; wave-butterfly stats, no barriers ---
#pragma unroll 1
  for (int rr = 0; rr < 2; ++rr) {
    const int row = wave * 2 + rr;
    const float* hrow = h + (size_t)(rowbase + row) * kD;
    float4 x[16];
    float s = 0.0f, ss = 0.0f;
#pragma unroll
    for (int j = 0; j < 16; ++j) {
      x[j] = *reinterpret_cast<const float4*>(hrow + (lane + j * 32) * 4);
      s  += x[j].x + x[j].y + x[j].z + x[j].w;
      ss += x[j].x * x[j].x + x[j].y * x[j].y + x[j].z * x[j].z +
            x[j].w * x[j].w;
    }
#pragma unroll
    for (int o = 16; o > 0; o >>= 1) {   // butterfly: all lanes get totals
      s  += __shfl_xor(s, o, 32);
      ss += __shfl_xor(ss, o, 32);
    }
    const float mu  = s * (1.0f / kD);
    const float var = fmaxf(ss * (1.0f / kD) - mu * mu, 0.0f);
    const float rs  = 1.0f / sqrtf(var + kEps);
    if (lane == 0) { sMu[row] = mu; sRs[row] = rs; }
    __bf16* arow = Ahat + row * kAStride;
#pragma unroll
    for (int j = 0; j < 16; ++j) {
      const int d = (lane + j * 32) * 4;
      arow[d + 0] = (__bf16)((x[j].x - mu) * rs);
      arow[d + 1] = (__bf16)((x[j].y - mu) * rs);
      arow[d + 2] = (__bf16)((x[j].z - mu) * rs);
      arow[d + 3] = (__bf16)((x[j].w - mu) * rs);
    }
  }
  __syncthreads();

  // --------- WMMA phase: C[16x16] = Ahat[16x2048] x V^T[2048x16] -----------
  // A layout (contiguous k, as B): lanes 0-15 row M=lane, elems 0..15=K0..15;
  // lanes 16-31 row M=lane-16, elems 0..15=K16..31.
  // B layout (B KxN striping): lanes 0-15 col N=lane hold K0..15 in order;
  // lanes 16-31 col N=lane-16 hold K16..31. Cols >=8 duplicate V rows 0..7.
  {
    const int M    = lane & 15;
    const int aoff = (lane < 16) ? 0 : 16;
    const int nn   = lane & 7;           // duplicate, cols 8..15 never read
    const int kb   = (lane < 16) ? 0 : 16;
    f32x8 acc = {0.f, 0.f, 0.f, 0.f, 0.f, 0.f, 0.f, 0.f};
#pragma unroll
    for (int c = 0; c < 8; ++c) {
      const int k0 = (wave * 8 + c) * 32;  // each wave owns 8 K-chunks of 32
      const bf16x8 lo =
          *reinterpret_cast<const bf16x8*>(&Ahat[M * kAStride + k0 + aoff]);
      const bf16x8 hi =
          *reinterpret_cast<const bf16x8*>(&Ahat[M * kAStride + k0 + aoff + 8]);
      const bf16x16 afrag = __builtin_shufflevector(
          lo, hi, 0, 1, 2, 3, 4, 5, 6, 7, 8, 9, 10, 11, 12, 13, 14, 15);
      const float4 f0 = *reinterpret_cast<const float4*>(V + nn * kD + k0 + kb);
      const float4 f1 =
          *reinterpret_cast<const float4*>(V + nn * kD + k0 + kb + 4);
      const float4 f2 =
          *reinterpret_cast<const float4*>(V + nn * kD + k0 + kb + 8);
      const float4 f3 =
          *reinterpret_cast<const float4*>(V + nn * kD + k0 + kb + 12);
      bf16x16 bfrag;
      bfrag[0]  = (__bf16)f0.x; bfrag[1]  = (__bf16)f0.y;
      bfrag[2]  = (__bf16)f0.z; bfrag[3]  = (__bf16)f0.w;
      bfrag[4]  = (__bf16)f1.x; bfrag[5]  = (__bf16)f1.y;
      bfrag[6]  = (__bf16)f1.z; bfrag[7]  = (__bf16)f1.w;
      bfrag[8]  = (__bf16)f2.x; bfrag[9]  = (__bf16)f2.y;
      bfrag[10] = (__bf16)f2.z; bfrag[11] = (__bf16)f2.w;
      bfrag[12] = (__bf16)f3.x; bfrag[13] = (__bf16)f3.y;
      bfrag[14] = (__bf16)f3.z; bfrag[15] = (__bf16)f3.w;
      acc = __builtin_amdgcn_wmma_f32_16x16x32_bf16(
          false, afrag, false, bfrag, (short)0, acc, false, false);
    }
    // C/D layout: lane<16 -> acc[i] = C[M=i][N=lane]; lane>=16 -> C[M=8+i][..]
    const int mbase = (lane < 16) ? 0 : 8;
    const int ncol  = lane & 15;
#pragma unroll
    for (int i = 0; i < 8; ++i) Cpart[wave][mbase + i][ncol] = acc[i];
  }
  __syncthreads();
  {
    const int m = tid >> 4, nc = tid & 15;
    float csum = 0.0f;
#pragma unroll
    for (int w = 0; w < kWaves; ++w) csum += Cpart[w][m][nc];
    sCoef[m][nc] = csum;
  }
  __syncthreads();

  // ------------- pass 2: proj, delta, e_curv, smoothed output --------------
  const int d0 = tid * 4, d1 = (tid + kBlock) * 4;
  const float4 g0 = *reinterpret_cast<const float4*>(gamma + d0);
  const float4 g1 = *reinterpret_cast<const float4*>(gamma + d1);
  const float4 b0 = *reinterpret_cast<const float4*>(beta + d0);
  const float4 b1 = *reinterpret_cast<const float4*>(beta + d1);

#pragma unroll 1
  for (int row = 0; row < kRows; ++row) {
    const float mu = sMu[row];
    const float rs = sRs[row];
    float cf[kRank];
#pragma unroll
    for (int rk = 0; rk < kRank; ++rk) cf[rk] = sCoef[row][rk];

    const size_t rowIdx = (size_t)(rowbase + row);
    const float* hrow = h + rowIdx * kD;  // L2-resident re-read
    const float4 x0 = *reinterpret_cast<const float4*>(hrow + d0);
    const float4 x1 = *reinterpret_cast<const float4*>(hrow + d1);
    const float4 hh0 = make_float4((x0.x - mu) * rs, (x0.y - mu) * rs,
                                   (x0.z - mu) * rs, (x0.w - mu) * rs);
    const float4 hh1 = make_float4((x1.x - mu) * rs, (x1.y - mu) * rs,
                                   (x1.z - mu) * rs, (x1.w - mu) * rs);
    float4 p0 = make_float4(0.f, 0.f, 0.f, 0.f);
    float4 p1 = make_float4(0.f, 0.f, 0.f, 0.f);
#pragma unroll
    for (int rk = 0; rk < kRank; ++rk) {
      const float  c  = cf[rk];
      const float4 v0 = *reinterpret_cast<const float4*>(V + rk * kD + d0);
      const float4 v1 = *reinterpret_cast<const float4*>(V + rk * kD + d1);
      p0.x += c * v0.x; p0.y += c * v0.y; p0.z += c * v0.z; p0.w += c * v0.w;
      p1.x += c * v1.x; p1.y += c * v1.y; p1.z += c * v1.z; p1.w += c * v1.w;
    }
    const float4 dl0 = make_float4(hh0.x - p0.x, hh0.y - p0.y,
                                   hh0.z - p0.z, hh0.w - p0.w);
    const float4 dl1 = make_float4(hh1.x - p1.x, hh1.y - p1.y,
                                   hh1.z - p1.z, hh1.w - p1.w);
    float ep = dl0.x * dl0.x + dl0.y * dl0.y + dl0.z * dl0.z + dl0.w * dl0.w +
               dl1.x * dl1.x + dl1.y * dl1.y + dl1.z * dl1.z + dl1.w * dl1.w;
    const float4 o0 = make_float4((hh0.x - etac * dl0.x) * g0.x + b0.x,
                                  (hh0.y - etac * dl0.y) * g0.y + b0.y,
                                  (hh0.z - etac * dl0.z) * g0.z + b0.z,
                                  (hh0.w - etac * dl0.w) * g0.w + b0.w);
    const float4 o1 = make_float4((hh1.x - etac * dl1.x) * g1.x + b1.x,
                                  (hh1.y - etac * dl1.y) * g1.y + b1.y,
                                  (hh1.z - etac * dl1.z) * g1.z + b1.z,
                                  (hh1.w - etac * dl1.w) * g1.w + b1.w);
    float* orow = out + rowIdx * kD;
    *reinterpret_cast<float4*>(orow + d0) = o0;
    *reinterpret_cast<float4*>(orow + d1) = o1;
    // wave-level e reduction only; block combine deferred past the loop
#pragma unroll
    for (int o = 16; o > 0; o >>= 1) ep += __shfl_xor(ep, o, 32);
    if (lane == 0) Epart[row][wave] = ep;
  }
  __syncthreads();
  if (tid < kRows) {
    float e = 0.0f;
#pragma unroll
    for (int w = 0; w < kWaves; ++w) e += Epart[tid][w];
    eout[rowbase + tid] = e;
  }
}

extern "C" void kernel_launch(void* const* d_in, const int* in_sizes, int n_in,
                              void* d_out, int out_size, void* d_ws,
                              size_t ws_size, hipStream_t stream) {
  (void)n_in; (void)out_size; (void)d_ws; (void)ws_size;
  const float* h     = (const float*)d_in[0];
  const float* gamma = (const float*)d_in[1];
  const float* beta  = (const float*)d_in[2];
  const float* V     = (const float*)d_in[3];
  const float* eta   = (const float*)d_in[4];
  float* out = (float*)d_out;
  const int nrows = in_sizes[0] / kD;                 // B*S = 16384
  float* eout = out + (size_t)nrows * kD;             // e_curv appended
  dim3 grid(nrows / kRows);
  lbonorm_kernel<<<grid, kBlock, 0, stream>>>(h, gamma, beta, V, eta, out,
                                              eout);
}
